// GQAAttention_29721173688520
// MI455X (gfx1250) — compile-verified, matrix operand order checked by reading, not run on hardware
//
#include <hip/hip_runtime.h>

// ---------------- problem constants ----------------
constexpr int B_   = 2;
constexpr int S_   = 2048;
constexpr int D_   = 2048;
constexpr int H_   = 16;
constexpr int KVH_ = 4;
constexpr int HD_  = 128;
constexpr int BS_  = B_ * S_;          // 4096 rows
constexpr int NQKV = H_ * HD_ + 2 * KVH_ * HD_; // 3072

typedef __attribute__((ext_vector_type(16))) __bf16 v16bf;
typedef __attribute__((ext_vector_type(8)))  float  v8f;

// ---------------- bf16 helpers (raw ushort storage) ----------------
__device__ __forceinline__ unsigned short f2bf(float x) {
  union { float f; unsigned u; } c; c.f = x;
  unsigned r = c.u + 0x7FFFu + ((c.u >> 16) & 1u);
  return (unsigned short)(r >> 16);
}
__device__ __forceinline__ float b2f(unsigned short h) {
  union { unsigned u; float f; } c; c.u = ((unsigned)h) << 16;
  return c.f;
}

// ---------------- CDNA5 async global->LDS copy (16 bytes / lane) ----------------
// GLOBAL_LOAD_ASYNC_TO_LDS_B128: vdst = VGPR holding LDS byte address,
// vaddr = 64-bit global address. Tracked with ASYNCcnt.
__device__ __forceinline__ void async_cp16(const unsigned short* g, unsigned short* l) {
  unsigned long long ga = (unsigned long long)(uintptr_t)g;
  unsigned la = (unsigned)(uintptr_t)l;   // generic LDS ptr: low 32 bits = LDS offset
  asm volatile("global_load_async_to_lds_b128 %0, %1, off"
               :: "v"(la), "v"(ga) : "memory");
}
__device__ __forceinline__ void async_wait0() {
  asm volatile("s_wait_asynccnt 0" ::: "memory");
}

// A-fragment (16x32 bf16) load from an LDS tile stored row-major.
// ISA layout: lanes 0-15 row M=lane, K 0..7 & 16..23; lanes 16-31 same rows, K 8..15 & 24..31.
// pitch must be a multiple of 8 elements (16B alignment for b128 LDS ops).
__device__ __forceinline__ v16bf load_frag(const unsigned short* base, int row0,
                                           int pitch, int kstep, int lane) {
  const int half = lane >> 4, l15 = lane & 15;
  const unsigned short* p = base + (size_t)(row0 + l15) * pitch + kstep + half * 8;
  union { uint4 u[2]; v16bf v; } f;
  f.u[0] = *(const uint4*)(p);
  f.u[1] = *(const uint4*)(p + 16);
  return f.v;
}

// ---------------- converts ----------------
__global__ void k_f32_to_bf16(const float* __restrict__ in,
                              unsigned short* __restrict__ out, int n) {
  int i = blockIdx.x * blockDim.x + threadIdx.x;
  if (i < n) out[i] = f2bf(in[i]);
}

// fuse Wq|Wk|Wv into one [D, 3072] bf16 matrix
__global__ void k_build_wqkv(const float* __restrict__ Wq, const float* __restrict__ Wk,
                             const float* __restrict__ Wv, unsigned short* __restrict__ out) {
  int i = blockIdx.x * blockDim.x + threadIdx.x;
  if (i >= D_ * NQKV) return;
  int k = i / NQKV, n = i % NQKV;
  float v;
  if (n < H_ * HD_)                 v = Wq[(size_t)k * (H_ * HD_) + n];
  else if (n < H_ * HD_ + KVH_*HD_) v = Wk[(size_t)k * (KVH_ * HD_) + (n - H_ * HD_)];
  else                              v = Wv[(size_t)k * (KVH_ * HD_) + (n - H_ * HD_ - KVH_ * HD_)];
  out[i] = f2bf(v);
}

// ---------------- bf16 WMMA GEMM: C[M,N] = A[M,K] * B[K,N] ----------------
// block tile 128x128, BK=32, 256 threads = 8 waves (2x4), each wave 64x32 (4x2 WMMA tiles)
template <bool OUT_BF16>
__global__ __launch_bounds__(256)
void k_gemm_bf16(const unsigned short* __restrict__ A, const unsigned short* __restrict__ Bm,
                 void* __restrict__ Cout, int M, int N, int K) {
  extern __shared__ char smem[];
  unsigned short* As = (unsigned short*)smem;   // [128][48] (pitch 48 -> 96B rows)
  unsigned short* Bs = As + 128 * 48;           // [128][48], holds B^T tile (N-major)

  const int tid = threadIdx.x;
  const int lane = tid & 31, wid = tid >> 5;
  const int wm = wid >> 2, wn = wid & 3;
  const int m0 = blockIdx.y * 128, n0 = blockIdx.x * 128;

  const v8f vz = {0.f,0.f,0.f,0.f,0.f,0.f,0.f,0.f};
  v8f acc[4][2];
#pragma unroll
  for (int im = 0; im < 4; im++)
#pragma unroll
    for (int in = 0; in < 2; in++) acc[im][in] = vz;

  const int arow = tid >> 1, acol = (tid & 1) * 16;   // A: 16 elems/thread (contiguous)
  const int bk = tid >> 3,  bn  = (tid & 7) * 16;     // B: 16 elems/thread

  for (int k0 = 0; k0 < K; k0 += 32) {
    __syncthreads();
    { // A tile -> LDS via async DMA (contiguous 32B per thread on both sides)
      const unsigned short* g = A + (size_t)(m0 + arow) * K + k0 + acol;
      unsigned short* d = As + arow * 48 + acol;
      async_cp16(g, d);
      async_cp16(g + 8, d + 8);
    }
    { // B tile -> LDS transposed (B^T) through registers
      const unsigned short* g = Bm + (size_t)(k0 + bk) * N + n0 + bn;
      unsigned short tmp[16];
      *(uint4*)(tmp)     = *(const uint4*)(g);
      *(uint4*)(tmp + 8) = *(const uint4*)(g + 8);
#pragma unroll
      for (int i = 0; i < 16; i++) Bs[(bn + i) * 48 + bk] = tmp[i];
    }
    if (k0 + 32 < K) {
      __builtin_prefetch(A + (size_t)(m0 + arow) * K + k0 + 32 + acol, 0, 1);
      __builtin_prefetch(Bm + (size_t)(k0 + 32 + bk) * N + n0 + bn, 0, 1);
    }
    async_wait0();
    __syncthreads();

    v16bf bfr[2];
#pragma unroll
    for (int in = 0; in < 2; in++) bfr[in] = load_frag(Bs, wn * 32 + in * 16, 48, 0, lane);
#pragma unroll
    for (int im = 0; im < 4; im++) {
      v16bf a = load_frag(As, wm * 64 + im * 16, 48, 0, lane);
#pragma unroll
      for (int in = 0; in < 2; in++)
        acc[im][in] = __builtin_amdgcn_wmma_f32_16x16x32_bf16(
            false, a, false, bfr[in], (short)0, acc[im][in], false, false);
    }
  }

  const int half = lane >> 4, l15 = lane & 15;
#pragma unroll
  for (int im = 0; im < 4; im++)
#pragma unroll
    for (int in = 0; in < 2; in++)
#pragma unroll
      for (int r = 0; r < 8; r++) {
        int m = m0 + wm * 64 + im * 16 + r + 8 * half;
        int n = n0 + wn * 32 + in * 16 + l15;
        float v = acc[im][in][r];
        if (OUT_BF16) ((unsigned short*)Cout)[(size_t)m * N + n] = f2bf(v);
        else          ((float*)Cout)[(size_t)m * N + n] = v;
      }
}

// ---------------- RoPE + head split ----------------
// qkv: [BS][3072] bf16 -> qh [B][H][S][HD], kh/vh [B][KVH][S][HD]  (bf16)
__global__ void k_rope_split(const unsigned short* __restrict__ qkv,
                             unsigned short* __restrict__ qh,
                             unsigned short* __restrict__ kh,
                             unsigned short* __restrict__ vh) {
  int i = blockIdx.x * blockDim.x + threadIdx.x;
  if (i >= BS_ * NQKV) return;
  int tok = i / NQKV, c = i % NQKV;
  int b = tok / S_, spos = tok % S_;
  const unsigned short* row = qkv + (size_t)tok * NQKV;

  if (c < H_ * HD_) {                          // Q + RoPE
    int head = c >> 7, d = c & 127, d2 = d & 63;
    float invf = powf(10000.0f, -(float)(2 * d2) * (1.0f / 128.0f));
    float ang = (float)spos * invf;
    float cs = cosf(ang), sn = sinf(ang);
    const unsigned short* hb = row + head * HD_;
    float x0 = b2f(hb[d2]), x1 = b2f(hb[d2 + 64]);
    float v = (d < 64) ? (x0 * cs - x1 * sn) : (x1 * cs + x0 * sn);
    qh[(((size_t)b * H_ + head) * S_ + spos) * HD_ + d] = f2bf(v);
  } else if (c < H_ * HD_ + KVH_ * HD_) {      // K + RoPE
    int cc = c - H_ * HD_;
    int head = cc >> 7, d = cc & 127, d2 = d & 63;
    float invf = powf(10000.0f, -(float)(2 * d2) * (1.0f / 128.0f));
    float ang = (float)spos * invf;
    float cs = cosf(ang), sn = sinf(ang);
    const unsigned short* hb = row + H_ * HD_ + head * HD_;
    float x0 = b2f(hb[d2]), x1 = b2f(hb[d2 + 64]);
    float v = (d < 64) ? (x0 * cs - x1 * sn) : (x1 * cs + x0 * sn);
    kh[(((size_t)b * KVH_ + head) * S_ + spos) * HD_ + d] = f2bf(v);
  } else {                                     // V passthrough
    int cc = c - H_ * HD_ - KVH_ * HD_;
    int head = cc >> 7, d = cc & 127;
    vh[(((size_t)b * KVH_ + head) * S_ + spos) * HD_ + d] = row[c];
  }
}

// ---------------- flash attention (causal, GQA) ----------------
// grid: (S/128, B*H); 256 thr = 8 waves; each wave owns 16 query rows.
// LDS: Q 128x136, K 64x136, V^T 128x72, P 8x(16x72)  => 89,088 bytes
__global__ __launch_bounds__(256)
void k_attn(const unsigned short* __restrict__ qh, const unsigned short* __restrict__ kh,
            const unsigned short* __restrict__ vh, unsigned short* __restrict__ attn) {
  extern __shared__ char smem[];
  unsigned short* Qs  = (unsigned short*)smem;       // pitch 136 (272B rows)
  unsigned short* Ks  = Qs + 128 * 136;              // pitch 136
  unsigned short* Vts = Ks + 64 * 136;               // pitch 72 (144B rows), V transposed
  unsigned short* Ps  = Vts + 128 * 72;              // per-wave 16x72

  const int tid = threadIdx.x, lane = tid & 31, wid = tid >> 5;
  const int half = lane >> 4, l15 = lane & 15;
  const int q0 = blockIdx.x * 128;
  const int bh = blockIdx.y;
  const int b = bh / H_, h = bh % H_, kvh = h >> 2;   // REP = 4

  const unsigned short* Qg = qh + (((size_t)b * H_ + h) * S_ + q0) * HD_;
  const unsigned short* Kg = kh + (((size_t)b * KVH_ + kvh) * S_) * HD_;
  const unsigned short* Vg = vh + (((size_t)b * KVH_ + kvh) * S_) * HD_;

  { // Q tile -> LDS via async DMA (contiguous 128B per thread on both sides)
    int row = tid >> 1, cb = (tid & 1) * 64;
    const unsigned short* g = Qg + (size_t)row * HD_ + cb;
    unsigned short* d = Qs + row * 136 + cb;
#pragma unroll
    for (int i = 0; i < 8; i++) async_cp16(g + 8 * i, d + 8 * i);
  }

  const v8f vz = {0.f,0.f,0.f,0.f,0.f,0.f,0.f,0.f};
  v8f oacc[8];
#pragma unroll
  for (int on = 0; on < 8; on++) oacc[on] = vz;
  float mrow[8], lrow[8];
#pragma unroll
  for (int r = 0; r < 8; r++) { mrow[r] = -1e30f; lrow[r] = 0.f; }

  const float sc = 0.08838834764831845f * 1.4426950408889634f; // (1/sqrt(128))*log2(e)

  const int kbmax = (q0 + 127) >> 6;   // causal: only KV blocks <= last query row
  for (int kb = 0; kb <= kbmax; kb++) {
    __syncthreads();
    { // K tile 64x128 -> LDS via async DMA (contiguous 64B per thread on both sides)
      int row = tid >> 2, cb = (tid & 3) * 32;
      const unsigned short* g = Kg + (size_t)(kb * 64 + row) * HD_ + cb;
      unsigned short* d = Ks + row * 136 + cb;
#pragma unroll
      for (int i = 0; i < 4; i++) async_cp16(g + 8 * i, d + 8 * i);
    }
    { // V tile, transposed into LDS (V^T: hd-major) for PV B-fragments
      int row = tid >> 2, cb = (tid & 3) * 32;
      const uint4* g = (const uint4*)(Vg + (size_t)(kb * 64 + row) * HD_ + cb);
      unsigned short tmp[32];
#pragma unroll
      for (int i = 0; i < 4; i++) *(uint4*)(tmp + 8 * i) = g[i];
#pragma unroll
      for (int i = 0; i < 32; i++) Vts[(cb + i) * 72 + row] = tmp[i];
    }
    async_wait0();      // covers Q tile (first iter) + K tile async DMAs
    __syncthreads();

    // S = Q @ K^T   (16 q-rows x 64 kv-cols per wave)
    v8f sacc[4] = {vz, vz, vz, vz};
#pragma unroll
    for (int ks = 0; ks < 4; ks++) {
      v16bf a = load_frag(Qs, wid * 16, 136, ks * 32, lane);
#pragma unroll
      for (int jn = 0; jn < 4; jn++) {
        v16bf bm = load_frag(Ks, jn * 16, 136, ks * 32, lane);
        sacc[jn] = __builtin_amdgcn_wmma_f32_16x16x32_bf16(
            false, a, false, bm, (short)0, sacc[jn], false, false);
      }
    }

    // online softmax: rows live at (vgpr r, lane-half); reduce across 16 lanes
#pragma unroll
    for (int r = 0; r < 8; r++) {
      const int qrg = q0 + wid * 16 + r + 8 * half;
      float mx = -1e30f;
#pragma unroll
      for (int jn = 0; jn < 4; jn++) {
        int kg = kb * 64 + jn * 16 + l15;
        float s = sacc[jn][r];
        if (kg > qrg) s = -1e30f;          // causal mask
        sacc[jn][r] = s;
        mx = fmaxf(mx, s);
      }
      mx = fmaxf(mx, __shfl_xor(mx, 1));
      mx = fmaxf(mx, __shfl_xor(mx, 2));
      mx = fmaxf(mx, __shfl_xor(mx, 4));
      mx = fmaxf(mx, __shfl_xor(mx, 8));
      float mnew = fmaxf(mrow[r], mx);
      float alpha = exp2f((mrow[r] - mnew) * sc);
      mrow[r] = mnew;
      float sum = 0.f;
#pragma unroll
      for (int jn = 0; jn < 4; jn++) {
        float p = exp2f((sacc[jn][r] - mnew) * sc);
        sacc[jn][r] = p;
        sum += p;
      }
      sum += __shfl_xor(sum, 1);
      sum += __shfl_xor(sum, 2);
      sum += __shfl_xor(sum, 4);
      sum += __shfl_xor(sum, 8);
      lrow[r] = lrow[r] * alpha + sum;
#pragma unroll
      for (int on = 0; on < 8; on++) oacc[on][r] *= alpha;
    }

    // P (C-layout fp32) -> per-wave LDS tile as bf16 A-operand
    unsigned short* Pw = Ps + wid * 16 * 72;
#pragma unroll
    for (int jn = 0; jn < 4; jn++)
#pragma unroll
      for (int r = 0; r < 8; r++)
        Pw[(r + 8 * half) * 72 + jn * 16 + l15] = f2bf(sacc[jn][r]);
    asm volatile("s_wait_dscnt 0" ::: "memory");   // wave-local LDS RAW fence

    // O += P @ V
#pragma unroll
    for (int ks = 0; ks < 2; ks++) {
      v16bf a = load_frag(Pw, 0, 72, ks * 32, lane);
#pragma unroll
      for (int on = 0; on < 8; on++) {
        v16bf bm = load_frag(Vts, on * 16, 72, ks * 32, lane);
        oacc[on] = __builtin_amdgcn_wmma_f32_16x16x32_bf16(
            false, a, false, bm, (short)0, oacc[on], false, false);
      }
    }
  }

  // epilogue: O /= l, write [B][S][H*HD] bf16
#pragma unroll
  for (int on = 0; on < 8; on++)
#pragma unroll
    for (int r = 0; r < 8; r++) {
      int row = wid * 16 + r + 8 * half;
      float v = oacc[on][r] / lrow[r];
      attn[((size_t)b * S_ + q0 + row) * (H_ * HD_) + h * HD_ + on * 16 + l15] = f2bf(v);
    }
}

// ---------------- host launch ----------------
extern "C" void kernel_launch(void* const* d_in, const int* in_sizes, int n_in,
                              void* d_out, int out_size, void* d_ws, size_t ws_size,
                              hipStream_t stream) {
  const float* x  = (const float*)d_in[0];
  const float* Wq = (const float*)d_in[1];
  const float* Wk = (const float*)d_in[2];
  const float* Wv = (const float*)d_in[3];
  const float* Wo = (const float*)d_in[4];
  float* out = (float*)d_out;

  char* ws = (char*)d_ws;
  size_t off = 0;
  auto carve = [&](size_t bytes) -> void* {
    void* p = ws + off;
    off += (bytes + 255) & ~(size_t)255;
    return p;
  };
  unsigned short* xb   = (unsigned short*)carve((size_t)BS_ * D_ * 2);
  unsigned short* wqkv = (unsigned short*)carve((size_t)D_ * NQKV * 2);
  unsigned short* wob  = (unsigned short*)carve((size_t)D_ * D_ * 2);
  unsigned short* qkv  = (unsigned short*)carve((size_t)BS_ * NQKV * 2);
  unsigned short* qhb  = (unsigned short*)carve((size_t)B_ * H_ * S_ * HD_ * 2);
  unsigned short* khb  = (unsigned short*)carve((size_t)B_ * KVH_ * S_ * HD_ * 2);
  unsigned short* vhb  = (unsigned short*)carve((size_t)B_ * KVH_ * S_ * HD_ * 2);
  unsigned short* attn = (unsigned short*)carve((size_t)BS_ * H_ * HD_ * 2);

  const int T = 256;
  // 1) converts
  {
    int n = BS_ * D_;
    k_f32_to_bf16<<<(n + T - 1) / T, T, 0, stream>>>(x, xb, n);
  }
  {
    int n = D_ * NQKV;
    k_build_wqkv<<<(n + T - 1) / T, T, 0, stream>>>(Wq, Wk, Wv, wqkv);
  }
  {
    int n = D_ * D_;
    k_f32_to_bf16<<<(n + T - 1) / T, T, 0, stream>>>(Wo, wob, n);
  }
  // 2) QKV projection GEMM: [4096,2048] x [2048,3072] -> bf16 qkv
  {
    dim3 grid(NQKV / 128, BS_ / 128);
    k_gemm_bf16<true><<<grid, T, 2 * 128 * 48 * 2, stream>>>(xb, wqkv, qkv, BS_, NQKV, D_);
  }
  // 3) RoPE + head split
  {
    int n = BS_ * NQKV;
    k_rope_split<<<(n + T - 1) / T, T, 0, stream>>>(qkv, qhb, khb, vhb);
  }
  // 4) causal GQA flash attention
  {
    dim3 grid(S_ / 128, B_ * H_);
    size_t lds = (size_t)(128 * 136 + 64 * 136 + 128 * 72 + 8 * 16 * 72) * 2;
    k_attn<<<grid, T, lds, stream>>>(qhb, khb, vhb, attn);
  }
  // 5) output projection GEMM: [4096,2048] x [2048,2048] -> fp32 out
  {
    dim3 grid(D_ / 128, BS_ / 128);
    k_gemm_bf16<false><<<grid, T, 2 * 128 * 48 * 2, stream>>>(attn, wob, out, BS_, D_, D_);
  }
}
